// SelfAttention_37787122270731
// MI455X (gfx1250) — matrix operand order repaired and checked by reading, not compiled, not run on hardware
//
#include <hip/hip_runtime.h>
#include <hip/hip_bf16.h>

// ---------------------------------------------------------------------------
// MI455X (gfx1250) fused self-attention:
//   h = x@W^T + b                          (bf16 WMMA, f32 accum)
//   out = tanh(softmax(tanh(h h^T)) @ h)   (flash-style, S never materialized)
// tanh bounds scores to [-1,1] => exp() needs no running max, so softmax
// fuses into one streaming pass with unnormalized accumulation + row sums.
// h / h^T (bf16, 32MB total) stay L2-resident (192MB L2) for the O(N^2) pass.
// ---------------------------------------------------------------------------

typedef __attribute__((ext_vector_type(16))) __bf16        v16bf;
typedef __attribute__((ext_vector_type(8)))  __bf16        v8bf;
typedef __attribute__((ext_vector_type(4)))  __bf16        v4bf;
typedef __attribute__((ext_vector_type(8)))  float         v8f;
typedef __attribute__((ext_vector_type(4)))  float         f32x4;
typedef __attribute__((ext_vector_type(4)))  unsigned int  u32x4;

// Exact pointee type the async-to-LDS builtin expects (per hipcc diagnostic):
// GCC-style vector_size(16) int, in addrspace(1) (src) / addrspace(3) (dst).
typedef int b128_t __attribute__((vector_size(16)));
typedef __attribute__((address_space(1))) b128_t* gptr128;
typedef __attribute__((address_space(3))) b128_t* lptr128;

#define NROWS 8192
#define DIM   1024
#define QLD   1032   // padded LDS stride for Q (bf16 elems; 2064B, 16B-aligned)
#define PLD   136    // padded LDS stride for P (bf16 elems; 272B, 16B-aligned)

__device__ __forceinline__ v8f wmma_bf16(v16bf a, v16bf b, v8f c) {
  // (neg_a, A, neg_b, B, c_mod, C, reuse_a, reuse_b)
  return __builtin_amdgcn_wmma_f32_16x16x32_bf16(false, a, false, b,
                                                 (short)0, c, false, false);
}

// Fast, correctly-saturating tanh: one v_exp_f32 + a few VALU.
__device__ __forceinline__ float tanh_fast(float x) {
  return 1.f - 2.f / (__expf(2.f * x) + 1.f);
}

// A operand (16x32 bf16) from row-major memory, row m0+lane%16.
// Lanes 0-15: K = kb+[0..7], kb+[16..23]; lanes 16-31: +8.
__device__ __forceinline__ v16bf load_A(const __bf16* base, int ld, int m0,
                                        int kb, int lane) {
  const int row  = m0 + (lane & 15);
  const int koff = (lane < 16) ? 0 : 8;
  const __bf16* p = base + (size_t)row * ld + kb + koff;
  union { v16bf v; u32x4 q[2]; } u;
  u.q[0] = *(const u32x4*)(p);
  u.q[1] = *(const u32x4*)(p + 16);
  return u.v;
}

// Same A layout but from an LDS-resident tile (row 0..15, padded stride).
__device__ __forceinline__ v16bf load_A_lds(const __bf16* base, int ld, int kb,
                                            int lane) {
  const int row  = lane & 15;
  const int koff = (lane < 16) ? 0 : 8;
  const __bf16* p = base + row * ld + kb + koff;
  union { v16bf v; u32x4 q[2]; } u;
  u.q[0] = *(const u32x4*)(p);
  u.q[1] = *(const u32x4*)(p + 16);
  return u.v;
}

// B operand (32x16 bf16): B[k][n] = base[(n0+lane%16)*ld + kb + k].
// Lanes 0-15: K = kb+[0..7], kb+[16..23]; lanes 16-31: +8.
__device__ __forceinline__ v16bf load_B(const __bf16* base, int ld, int n0,
                                        int kb, int lane) {
  const int col  = n0 + (lane & 15);
  const int koff = (lane < 16) ? 0 : 8;
  const __bf16* p = base + (size_t)col * ld + kb + koff;
  union { v16bf v; u32x4 q[2]; } u;
  u.q[0] = *(const u32x4*)(p);
  u.q[1] = *(const u32x4*)(p + 16);
  return u.v;
}

// ---------------------------------------------------------------------------
// Kernel 0: f32 -> bf16 conversion of x and W (4-wide vectorized).
// ---------------------------------------------------------------------------
__global__ void cvt_f32_bf16(const float* __restrict__ x,
                             const float* __restrict__ W,
                             __bf16* __restrict__ xb,
                             __bf16* __restrict__ Wb) {
  const long NX = (long)NROWS * DIM;
  const long NW = (long)DIM * DIM;
  long base = ((long)blockIdx.x * blockDim.x + threadIdx.x) * 4;
  if (base < NX) {
    f32x4 f = *(const f32x4*)(x + base);
    *(v4bf*)(xb + base) = __builtin_convertvector(f, v4bf);
  } else if (base < NX + NW) {
    f32x4 f = *(const f32x4*)(W + (base - NX));
    *(v4bf*)(Wb + (base - NX)) = __builtin_convertvector(f, v4bf);
  }
}

// ---------------------------------------------------------------------------
// Kernel 1: h = x @ W^T + b.  128x128 tile per workgroup, 8 waves as 4x2,
// each wave owns 32x64 (2x4 grid of 16x16 WMMA accumulators).
// Writes h (row-major) AND h^T.  The WMMA C layout gives each lane a column
// fragment (8 consecutive m at fixed n), which is CONTIGUOUS in h^T: those
// stores go out as one aligned 16B global_store_b128 per tile.
// ---------------------------------------------------------------------------
__global__ __launch_bounds__(256) void gemm_h(const __bf16* __restrict__ xb,
                                              const __bf16* __restrict__ Wb,
                                              const float*  __restrict__ bias,
                                              __bf16* __restrict__ hb,
                                              __bf16* __restrict__ hTb) {
  const int tid  = threadIdx.x;
  const int w    = tid >> 5;
  const int lane = tid & 31;
  const int m0 = blockIdx.x * 128 + (w & 3) * 32;   // 4 row groups
  const int n0 = blockIdx.y * 128 + (w >> 2) * 64;  // 2 col groups

  v8f acc[2][4] = {};
  for (int kb = 0; kb < DIM; kb += 32) {
    v16bf a[2], bm[4];
    a[0] = load_A(xb, DIM, m0,      kb, lane);
    a[1] = load_A(xb, DIM, m0 + 16, kb, lane);
#pragma unroll
    for (int j = 0; j < 4; ++j) bm[j] = load_B(Wb, DIM, n0 + 16 * j, kb, lane);
#pragma unroll
    for (int i = 0; i < 2; ++i)
#pragma unroll
      for (int j = 0; j < 4; ++j) acc[i][j] = wmma_bf16(a[i], bm[j], acc[i][j]);
  }

  const int col16 = lane & 15;
  const int rbase = (lane < 16) ? 0 : 8;  // C layout: VGPR v -> row v or v+8
#pragma unroll
  for (int j = 0; j < 4; ++j) {
    const int n = n0 + 16 * j + col16;
    const float bv = bias[n];
#pragma unroll
    for (int i = 0; i < 2; ++i) {
      const int mbase = m0 + 16 * i + rbase;  // 8 consecutive rows
      v8f hf = acc[i][j];
#pragma unroll
      for (int v = 0; v < 8; ++v) hf[v] += bv;
      const v8bf h8 = __builtin_convertvector(hf, v8bf);
      // h^T: column fragment is contiguous -> single 16B store (16B-aligned:
      // mbase is a multiple of 8 bf16 elements).
      *(v8bf*)(hTb + (size_t)n * NROWS + mbase) = h8;
      // h (row-major): inherently strided per lane -> elementwise stores.
#pragma unroll
      for (int v = 0; v < 8; ++v)
        hb[(size_t)(mbase + v) * DIM + n] = h8[v];
    }
  }
}

// ---------------------------------------------------------------------------
// Kernel 2: fused scores + softmax + attn@h.
// One workgroup (8 waves) per 16 output rows; j advances 128/step so each
// wave owns a private 16x16 score tile with the FULL K=1024 reduction:
// no cross-wave LDS reduction, no atomics, minimal transcendental work.
//   - Q (16x1024 bf16) staged in LDS once (async-to-LDS when available)
//   - per step: 32 S-WMMAs -> 8 exp(tanh) per lane -> publish bf16 P slice
//     -> 32 PV-WMMAs against this wave's 128-col slice of h^T (L2-resident)
//   - row sums ride in registers; one shfl+LDS reduce at the end
// ---------------------------------------------------------------------------
__global__ __launch_bounds__(256) void attn_fused(const __bf16* __restrict__ hb,
                                                  const __bf16* __restrict__ hTb,
                                                  float* __restrict__ out) {
  __shared__ __attribute__((aligned(16))) __bf16 Qs[16 * QLD];
  __shared__ __attribute__((aligned(16))) __bf16 Pt[16 * PLD];
  __shared__ float Rsum[16];

  const int tid   = threadIdx.x;
  const int w     = tid >> 5;
  const int lane  = tid & 31;
  const int i0    = blockIdx.x * 16;  // output row block
  const int n0    = w * 128;          // wave's output column slice
  const int col16 = lane & 15;
  const int rbase = (lane < 16) ? 0 : 8;

  if (tid < 16) Rsum[tid] = 0.f;

  // ---- Stage Q = h[i0:i0+16, 0:1024] into LDS (2048 x 16B chunks) ----
  for (int c = tid; c < 16 * 128; c += 256) {
    const int row = c >> 7, cc = c & 127;
    const __bf16* src = hb + (size_t)(i0 + row) * DIM + cc * 8;
    __bf16* dst = Qs + row * QLD + cc * 8;
#if __has_builtin(__builtin_amdgcn_global_load_async_to_lds_b128)
    __builtin_amdgcn_global_load_async_to_lds_b128(
        (gptr128)src, (lptr128)dst, 0, 0);
#else
    *(u32x4*)dst = *(const u32x4*)src;
#endif
  }
#if __has_builtin(__builtin_amdgcn_global_load_async_to_lds_b128)
#if __has_builtin(__builtin_amdgcn_s_wait_asynccnt)
  __builtin_amdgcn_s_wait_asynccnt(0);
#else
  asm volatile("s_wait_asynccnt 0" ::: "memory");
#endif
#endif
  __syncthreads();

  v8f   acc[8] = {};
  float rs[8]  = {};

  for (int jb = 0; jb < NROWS; jb += 128) {
    // 1. Private score tile: rows i0..i0+15, cols jb+w*16..+16, K = 1024.
    v8f c = {};
#pragma unroll 4
    for (int ks = 0; ks < 32; ++ks) {
      v16bf aq = load_A_lds(Qs, QLD, ks * 32, lane);
      v16bf bq = load_B(hb, DIM, jb + w * 16, ks * 32, lane);
      c = wmma_bf16(aq, bq, c);
    }
    if (jb + 128 < NROWS)  // pull next j-block of h^T toward the WGP
      __builtin_prefetch(hTb + (size_t)(n0 + col16) * NROWS + jb + 128, 0, 0);

    __syncthreads();  // previous Pt fully consumed by all waves

    // 2. p = exp(tanh(s)); publish bf16 P slice; accumulate private rowsums.
#pragma unroll
    for (int v = 0; v < 8; ++v) {
      const float p = __expf(tanh_fast(c[v]));
      rs[v] += p;
      Pt[(rbase + v) * PLD + w * 16 + col16] = (__bf16)p;
    }
    __syncthreads();

    // 3. acc[t] += P(16x128) @ h[jb:jb+128, n-slice]   (B from h^T rows).
#pragma unroll
    for (int ks = 0; ks < 4; ++ks) {
      v16bf pa = load_A_lds(Pt, PLD, ks * 32, lane);
#pragma unroll
      for (int t = 0; t < 8; ++t) {
        v16bf bv = load_B(hTb, NROWS, n0 + t * 16, jb + ks * 32, lane);
        acc[t] = wmma_bf16(pa, bv, acc[t]);
      }
    }
  }

  // ---- Row-sum reduction: 16 lanes of each half (xor 1,2,4,8), then
  //      across the 8 waves via LDS. ----
#pragma unroll
  for (int v = 0; v < 8; ++v)
#pragma unroll
    for (int off = 1; off < 16; off <<= 1)
      rs[v] += __shfl_xor(rs[v], off, 32);
  if (col16 == 0) {  // lanes 0 (rows 0..7) and 16 (rows 8..15) of each wave
#pragma unroll
    for (int v = 0; v < 8; ++v) atomicAdd(&Rsum[rbase + v], rs[v]);
  }
  __syncthreads();

  float rtot[8];
#pragma unroll
  for (int v = 0; v < 8; ++v) rtot[v] = Rsum[rbase + v];

#pragma unroll
  for (int t = 0; t < 8; ++t) {
    const int n = n0 + t * 16 + col16;
#pragma unroll
    for (int v = 0; v < 8; ++v) {
      const int m = i0 + rbase + v;
      out[(size_t)m * DIM + n] = tanh_fast(acc[t][v] / rtot[v]);
    }
  }
}

// ---------------------------------------------------------------------------
// Launch: cvt -> gemm_h -> attn_fused.  Workspace layout (50 MiB):
//   [0,16Mi): xb   [16Mi,18Mi): Wb   [18Mi,34Mi): h   [34Mi,50Mi): h^T
// ---------------------------------------------------------------------------
extern "C" void kernel_launch(void* const* d_in, const int* in_sizes, int n_in,
                              void* d_out, int out_size, void* d_ws, size_t ws_size,
                              hipStream_t stream) {
  const float* x = (const float*)d_in[0];
  const float* W = (const float*)d_in[1];
  const float* b = (const float*)d_in[2];
  float* out = (float*)d_out;

  char* ws = (char*)d_ws;
  __bf16* xb  = (__bf16*)(ws);
  __bf16* Wb  = (__bf16*)(ws + (size_t)16 * 1024 * 1024);
  __bf16* hb  = (__bf16*)(ws + (size_t)18 * 1024 * 1024);
  __bf16* hTb = (__bf16*)(ws + (size_t)34 * 1024 * 1024);

  const long nconv4 = ((long)NROWS * DIM + (long)DIM * DIM) / 4; // 2359296
  cvt_f32_bf16<<<(int)(nconv4 / 256), 256, 0, stream>>>(x, W, xb, Wb);
  gemm_h<<<dim3(NROWS / 128, DIM / 128), 256, 0, stream>>>(xb, Wb, b, hb, hTb);
  attn_fused<<<NROWS / 16, 256, 0, stream>>>(hb, hTb, out);
}
